// axis_aligned_gmm_cmm_layer_2671469658373
// MI455X (gfx1250) — matrix operand order repaired and checked by reading, not compiled, not run on hardware
//
#include <hip/hip_runtime.h>

typedef __attribute__((ext_vector_type(16))) _Float16 v16h;
typedef __attribute__((ext_vector_type(8)))  _Float16 v8h;
typedef __attribute__((ext_vector_type(8)))  float    v8f;
typedef __attribute__((ext_vector_type(4)))  float    v4f;

#define DIM   128
#define KC    512
#define BATCH 65536
#define BM    32          // rows per workgroup
#define AK    256         // GEMM reduction length
#define APAD  264         // padded A row (halves)
#define DPAD  520         // padded dist row (floats)
#define LN_2PI 1.83787706641f

union H16 { v16h v; v8h h[2]; };

__device__ __forceinline__ v8f wmma16(v16h a, v16h b, v8f c) {
  return __builtin_amdgcn_wmma_f32_16x16x32_f16(false, a, false, b, (short)0, c,
                                                false, false);
}

__device__ __forceinline__ void split_f16(float v, _Float16& hi, _Float16& lo) {
  hi = (_Float16)v;
  lo = (_Float16)(v - (float)hi);
}

// ---------------------------------------------------------------------------
// Kernel 1: build split-f16 B matrix [K][256] = [Dinv | -2*c*Dinv], plus
// log_det[k], c2[k] = sum c^2*Dinv.
// ---------------------------------------------------------------------------
__global__ __launch_bounds__(128) void prep_kernel(
    const float* __restrict__ centers, const float* __restrict__ Dp,
    _Float16* __restrict__ Bhi, _Float16* __restrict__ Blo,
    float* __restrict__ ldet, float* __restrict__ c2) {
  const int k = blockIdx.x;
  const int d = threadIdx.x;
  const float dabs = fabsf(Dp[k * DIM + d]) + 1e-8f;
  const float dinv = 1.0f / dabs;
  const float c    = centers[k * DIM + d];

  _Float16 hi, lo;
  split_f16(dinv, hi, lo);
  Bhi[k * AK + d] = hi;  Blo[k * AK + d] = lo;
  split_f16(-2.0f * c * dinv, hi, lo);
  Bhi[k * AK + DIM + d] = hi;  Blo[k * AK + DIM + d] = lo;

  __shared__ float redx[128];
  __shared__ float redy[128];
  redx[d] = __logf(dabs) * 0.0f + logf(dabs);
  redy[d] = c * c * dinv;
  __syncthreads();
  for (int s = 64; s > 0; s >>= 1) {
    if (d < s) { redx[d] += redx[d + s]; redy[d] += redy[d + s]; }
    __syncthreads();
  }
  if (d == 0) { ldet[k] = redx[0]; c2[k] = redy[0]; }
}

// ---------------------------------------------------------------------------
// Kernel 2: g[k] = -0.5*ldet[k] - max_k(-0.5*ldet[k])   (log det_scale_safe)
// ---------------------------------------------------------------------------
__global__ __launch_bounds__(512) void dsf_kernel(const float* __restrict__ ldet,
                                                  float* __restrict__ g) {
  const int k = threadIdx.x;
  const float s = -0.5f * ldet[k];
  __shared__ float red[512];
  red[k] = s;
  __syncthreads();
  for (int st = 256; st > 0; st >>= 1) {
    if (k < st) red[k] = fmaxf(red[k], red[k + st]);
    __syncthreads();
  }
  g[k] = s - red[0];
}

// ---------------------------------------------------------------------------
// Kernel 3: fused dist_sq GEMM (split-f16 WMMA) + gmm/cmm responsibilities +
// argmax/CE partial. One WG = 32 batch rows x all 512 components.
// ---------------------------------------------------------------------------
__global__ __launch_bounds__(256) void main_kernel(
    const float* __restrict__ x, const _Float16* __restrict__ Bhi,
    const _Float16* __restrict__ Blo, const float* __restrict__ ldet,
    const float* __restrict__ c2, const float* __restrict__ g,
    float* __restrict__ out, float* __restrict__ partials) {
  extern __shared__ char smem[];
  _Float16* Ahi  = (_Float16*)smem;                       // 32*264*2 = 16896
  _Float16* Alo  = (_Float16*)(smem + 16896);             // 16896
  float*    dist = (float*)(smem + 33792);                // 32*520*4 = 66560
  float*    gS   = (float*)(smem + 100352);               // 512
  float*    c2S  = gS + KC;                               // 512
  float*    ldS  = c2S + KC;                              // 512
  float*    rowCe = ldS + KC;                             // 32

  const int t = threadIdx.x;
  const int rowBase = blockIdx.x * BM;

  // ---- stage per-component scalars ----
  gS[t]        = g[t];        gS[t + 256]  = g[t + 256];
  c2S[t]       = c2[t];       c2S[t + 256] = c2[t + 256];
  ldS[t]       = ldet[t];     ldS[t + 256] = ldet[t + 256];

  // ---- stage split-f16 A tile: cols [0,128)=x^2, [128,256)=x ----
  {
    const int r = t >> 3, sub = t & 7;
    const float* xr = x + (size_t)(rowBase + r) * DIM;
#pragma unroll
    for (int q = 0; q < 4; ++q) {
      const int j = 4 * sub + 32 * q;
      v4f xv = *(const v4f*)(xr + j);
#pragma unroll
      for (int e = 0; e < 4; ++e) {
        const float v = xv[e];
        _Float16 hi, lo;
        split_f16(v * v, hi, lo);
        Ahi[r * APAD + j + e] = hi;
        Alo[r * APAD + j + e] = lo;
        split_f16(v, hi, lo);
        Ahi[r * APAD + DIM + j + e] = hi;
        Alo[r * APAD + DIM + j + e] = lo;
      }
    }
  }
  __syncthreads();

  // ---- GEMM: wave w owns N-range [64w, 64w+64), both 16-row M halves ----
  {
    const int w = t >> 5;
    const int l = t & 31;
    const int m = l & 15;          // M (A) / N (B,C) lane coordinate
    const int hl = l >> 4;         // lane half

    v8f acc[8];
#pragma unroll
    for (int i = 0; i < 8; ++i) acc[i] = (v8f){0, 0, 0, 0, 0, 0, 0, 0};

    for (int c = 0; c < 8; ++c) {          // 8 chunks of K=32
      const int kb = 32 * c;
      const int o1 = kb + 8 * hl;          // ISA 16-bit A layout: VGPR0-3
      const int o2 = kb + 16 + 8 * hl;     //                      VGPR4-7
      H16 ah0, al0, ah1, al1;
      ah0.h[0] = *(const v8h*)&Ahi[m * APAD + o1];
      ah0.h[1] = *(const v8h*)&Ahi[m * APAD + o2];
      al0.h[0] = *(const v8h*)&Alo[m * APAD + o1];
      al0.h[1] = *(const v8h*)&Alo[m * APAD + o2];
      ah1.h[0] = *(const v8h*)&Ahi[(16 + m) * APAD + o1];
      ah1.h[1] = *(const v8h*)&Ahi[(16 + m) * APAD + o2];
      al1.h[0] = *(const v8h*)&Alo[(16 + m) * APAD + o1];
      al1.h[1] = *(const v8h*)&Alo[(16 + m) * APAD + o2];

#pragma unroll
      for (int tt = 0; tt < 4; ++tt) {     // 4 N-tiles of 16
        const int nrow = 64 * w + 16 * tt + m;      // B lane = column m
        const _Float16* bph = Bhi + (size_t)nrow * AK + kb + 8 * hl;
        const _Float16* bpl = Blo + (size_t)nrow * AK + kb + 8 * hl;
        H16 bh, bl;
        bh.h[0] = *(const v8h*)bph;  bh.h[1] = *(const v8h*)(bph + 16);
        bl.h[0] = *(const v8h*)bpl;  bl.h[1] = *(const v8h*)(bpl + 16);

        // split product: hi*hi + hi*lo + lo*hi  (~f32 accuracy)
        acc[2 * tt + 0] = wmma16(ah0.v, bh.v, acc[2 * tt + 0]);
        acc[2 * tt + 0] = wmma16(ah0.v, bl.v, acc[2 * tt + 0]);
        acc[2 * tt + 0] = wmma16(al0.v, bh.v, acc[2 * tt + 0]);
        acc[2 * tt + 1] = wmma16(ah1.v, bh.v, acc[2 * tt + 1]);
        acc[2 * tt + 1] = wmma16(ah1.v, bl.v, acc[2 * tt + 1]);
        acc[2 * tt + 1] = wmma16(al1.v, bh.v, acc[2 * tt + 1]);
      }
    }

    // scatter dist (+ c2) to LDS per ISA C/D layout: VGPR v -> M = v + 8*half
#pragma unroll
    for (int tt = 0; tt < 4; ++tt) {
      const int col = 64 * w + 16 * tt + m;
      const float cc = c2S[col];
#pragma unroll
      for (int h2 = 0; h2 < 2; ++h2) {
        v8f a = acc[2 * tt + h2];
#pragma unroll
        for (int v = 0; v < 8; ++v) {
          const int rloc = 16 * h2 + 8 * hl + v;
          dist[rloc * DPAD + col] = a[v] + cc;
        }
      }
    }
  }
  __syncthreads();

  // ---- fused responsibilities: 8 lanes per batch row ----
  {
    const int r = t >> 3, sub = t & 7;
    const float* dr = &dist[r * DPAD];

    // pass 1: row maxima + argmax of (dsf + expo_gmm)
    float mg = -3.4e38f, mc = -3.4e38f, av = -3.4e38f;
    int ai = 0;
    for (int i = 0; i < 64; ++i) {
      const int k = 8 * i + sub;
      const float d = dr[k];
      const float vg = -0.5f * d;
      const float vc = expf(-64.5f * log1pf(d));
      mg = fmaxf(mg, vg);
      mc = fmaxf(mc, vc);
      const float sg = gS[k] + vg;
      if (sg > av || (sg == av && k < ai)) { av = sg; ai = k; }
    }
    for (int s = 1; s < 8; s <<= 1) {
      mg = fmaxf(mg, __shfl_xor(mg, s, 8));
      mc = fmaxf(mc, __shfl_xor(mc, s, 8));
      const float ov = __shfl_xor(av, s, 8);
      const int   oi = __shfl_xor(ai, s, 8);
      if (ov > av || (ov == av && oi < ai)) { av = ov; ai = oi; }
    }

    // pass 2: denominators
    float deng = 0.0f, denc = 0.0f;
    for (int i = 0; i < 64; ++i) {
      const int k = 8 * i + sub;
      const float d = dr[k];
      deng += expf(gS[k] - 0.5f * d - mg);
      denc += expf(gS[k] + expf(-64.5f * log1pf(d)) - mc);
    }
    for (int s = 1; s < 8; s <<= 1) {
      deng += __shfl_xor(deng, s, 8);
      denc += __shfl_xor(denc, s, 8);
    }
    const float thg = 1e-8f * deng, thc = 1e-8f * denc;

    // pass 3: clipped-softmax normalizers
    float Sg = 0.0f, Sc = 0.0f;
    for (int i = 0; i < 64; ++i) {
      const int k = 8 * i + sub;
      const float d = dr[k];
      Sg += fmaxf(expf(gS[k] - 0.5f * d - mg), thg);
      Sc += fmaxf(expf(gS[k] + expf(-64.5f * log1pf(d)) - mc), thc);
    }
    for (int s = 1; s < 8; s <<= 1) {
      Sg += __shfl_xor(Sg, s, 8);
      Sc += __shfl_xor(Sc, s, 8);
    }
    const float iSg = 1.0f / Sg, iSc = 1.0f / Sc;

    // pass 4: resp = log((softmax(resp_gmm)+softmax(resp_cmm))/2)
    float* outr = out + (size_t)(rowBase + r) * KC;
    for (int i = 0; i < 64; ++i) {
      const int k = 8 * i + sub;
      const float d = dr[k];
      const float a = fmaxf(expf(gS[k] - 0.5f * d - mg), thg) * iSg;
      const float b = fmaxf(expf(gS[k] + expf(-64.5f * log1pf(d)) - mc), thc) * iSc;
      outr[k] = logf(0.5f * (a + b));
    }

    if (sub == 0)
      rowCe[r] = -0.5f * (float)DIM * LN_2PI + ldS[ai] - 0.5f * dr[ai];
  }
  __syncthreads();
  if (t == 0) {
    float s = 0.0f;
    for (int r = 0; r < BM; ++r) s += rowCe[r];   // fixed order: deterministic
    partials[blockIdx.x] = s;
  }
}

// ---------------------------------------------------------------------------
// Kernel 4: deterministic fixed-order sum of 2048 CE partials
// ---------------------------------------------------------------------------
__global__ __launch_bounds__(256) void ce_final_kernel(
    const float* __restrict__ partials, float* __restrict__ ce_out) {
  __shared__ float red[256];
  const int t = threadIdx.x;
  float s = 0.0f;
  for (int i = 0; i < 8; ++i) s += partials[t * 8 + i];
  red[t] = s;
  __syncthreads();
  for (int st = 128; st > 0; st >>= 1) {
    if (t < st) red[t] += red[t + st];
    __syncthreads();
  }
  if (t == 0) ce_out[0] = -red[0] / (float)BATCH;
}

// ---------------------------------------------------------------------------
extern "C" void kernel_launch(void* const* d_in, const int* in_sizes, int n_in,
                              void* d_out, int out_size, void* d_ws,
                              size_t ws_size, hipStream_t stream) {
  const float* x       = (const float*)d_in[0];
  const float* centers = (const float*)d_in[1];
  const float* Dp      = (const float*)d_in[2];
  float* out = (float*)d_out;

  char* ws = (char*)d_ws;
  _Float16* Bhi   = (_Float16*)ws;                    // 512*256*2 = 262144
  _Float16* Blo   = (_Float16*)(ws + 262144);         // 262144
  float* ldet     = (float*)(ws + 524288);            // 2048
  float* c2v      = (float*)(ws + 526336);            // 2048
  float* gv       = (float*)(ws + 528384);            // 2048
  float* partials = (float*)(ws + 530432);            // 8192

  prep_kernel<<<KC, 128, 0, stream>>>(centers, Dp, Bhi, Blo, ldet, c2v);
  dsf_kernel<<<1, KC, 0, stream>>>(ldet, gv);

  const size_t shmem = 100352 + 3 * KC * 4 + BM * 4;  // 106624 bytes
  main_kernel<<<BATCH / BM, 256, shmem, stream>>>(x, Bhi, Blo, ldet, c2v, gv,
                                                  out, partials);
  ce_final_kernel<<<1, 256, 0, stream>>>(partials, out + (size_t)BATCH * KC);
}
